// Qwen3MoeDecoderLayer_76647986365151
// MI455X (gfx1250) — matrix operand order repaired and checked by reading, not compiled, not run on hardware
//
#include <hip/hip_runtime.h>
#include <hip/hip_bf16.h>

typedef __bf16 bf16_t;
typedef __attribute__((ext_vector_type(16))) __bf16 v16bf;
typedef __attribute__((ext_vector_type(8)))  float  v8f;
typedef unsigned int v4u __attribute__((ext_vector_type(4)));
typedef int          v4i __attribute__((ext_vector_type(4)));
typedef int          v8i __attribute__((ext_vector_type(8)));

#define TT    2048
#define HH    2048
#define NHD   32
#define NKVH  4
#define HDIM  128
#define NEXP  64
#define II    768
#define TOPKK 8
#define QKVN  5120          // 32*128 + 2*4*128
#define NROWS (TT*TOPKK)    // 16384

#define BM 128
#define BN 128
#define BK 32

__device__ __forceinline__ float waveSum(float v) {
#pragma unroll
  for (int o = 16; o > 0; o >>= 1) v += __shfl_xor(v, o, 32);
  return v;
}
__device__ __forceinline__ float waveMax(float v) {
#pragma unroll
  for (int o = 16; o > 0; o >>= 1) v = fmaxf(v, __shfl_xor(v, o, 32));
  return v;
}

// ---------------------------------------------------------------------------
// TDM: DMA a dense 128-row x 32-element bf16 tile (row stride = strideElems)
// from global memory into contiguous LDS at ldsOff.  D# per CDNA5 ISA §8.
// Issued by one wave (EXEC ignored by TDM); completion via s_wait_tensorcnt.
// ---------------------------------------------------------------------------
__device__ __forceinline__ void tdm_tile_128x32(const bf16_t* gsrc,
                                                long strideElems,
                                                unsigned ldsOff)
{
  unsigned long long ga = (unsigned long long)(const void*)gsrc;
  v4u g0;
  g0.x = 1u;                                        // count=1 (valid), user mode
  g0.y = ldsOff;                                    // lds_addr (bytes)
  g0.z = (unsigned)(ga & 0xffffffffu);              // global_addr[31:0]
  g0.w = (unsigned)((ga >> 32) & 0x01ffffffu)       // global_addr[56:32]
       | (2u << 30);                                // type = 2 ("image")
  v8i g1;
  g1[0] = (1 << 16);                                // wg_mask=0, data_size=1 (2B)
  g1[1] = (BK & 0xffff) << 16;                      // tensor_dim0[15:0]
  g1[2] = (BM & 0xffff) << 16;                      // dim0 hi=0, tensor_dim1[15:0]
  g1[3] = (BK & 0xffff) << 16;                      // dim1 hi=0, tile_dim0=32
  g1[4] = (BM & 0xffff);                            // tile_dim1=128, tile_dim2=0
  g1[5] = (int)(strideElems & 0xffffffff);          // tensor_dim0_stride[31:0]
  g1[6] = (int)((strideElems >> 32) & 0xffff);      // stride[47:32], dim1_stride=0
  g1[7] = 0;
  v4i z4 = {0, 0, 0, 0};
#if __clang_major__ >= 23
  v8i z8 = {0, 0, 0, 0, 0, 0, 0, 0};
  __builtin_amdgcn_tensor_load_to_lds(g0, g1, z4, z4, z8, 0);
#else
  __builtin_amdgcn_tensor_load_to_lds(g0, g1, z4, z4, 0);
#endif
}

// ---------------------------------------------------------------------------
// Flexible WMMA GEMM: C(f32) = A(bf16, gathered/ragged) x B(f32-or-bf16)
// 128x128 tile, 256 threads (8 waves in 2x4), K-step 32 == WMMA K.
// Double-buffered pipeline: while the 8 WMMAs of K-step kt run, the next
// tile is in flight (TDM async DMA tracked by TENSORcnt, or VMEM loads whose
// ds_store -- and therefore s_wait_loadcnt -- lands after the WMMA block).
// epi==0: store C;  epi==1: atomicAdd(out[tok]) of gamma-scaled result.
// ---------------------------------------------------------------------------
__global__ __launch_bounds__(256) void k_gemm(
    const bf16_t* __restrict__ A, long sAm,
    const float*  __restrict__ Bf, const bf16_t* __restrict__ Bb,
    long sBk, long sBn, long bExpStride,
    float* __restrict__ C, long sCm,
    int M, int N, int K,
    const int* __restrict__ aRowIdx,
    const int* __restrict__ grpOff, const int* __restrict__ grpCnt,
    const float* __restrict__ rowGamma, const int* __restrict__ epiTok,
    float* __restrict__ outAcc, long outStride, int epi)
{
  __shared__ bf16_t As[2][BM * BK];   // [m][k], ping-pong
  __shared__ bf16_t Bs[2][BN * BK];   // [n][k], ping-pong

  int e = blockIdx.z;
  int rowBase = 0, cnt = M;
  if (grpOff) { rowBase = grpOff[e]; cnt = grpCnt[e]; }
  int mTile = blockIdx.y * BM;
  if (mTile >= cnt) return;
  int nTile = blockIdx.x * BN;

  const float*  Bfe = Bf ? (Bf + (long)e * bExpStride) : (const float*)0;
  const bf16_t* Bbe = Bb ? (Bb + (long)e * bExpStride) : (const bf16_t*)0;

  int tid  = threadIdx.x;
  int lane = tid & 31;
  int wave = tid >> 5;
  int wm = (wave >> 2) * 64;
  int wn = (wave & 3) * 32;
  int lm    = lane & 15;
  int aKsel = (lane >> 4) << 3;    // 0 or 8  (hi lane half: A dword offset, K-offset 16)
  int bKsel = (lane >> 4) << 4;    // 0 or 16 (B K-offset for hi lane half)

  bool aTdm = (aRowIdx == nullptr) && (grpOff == nullptr);
  bool bTdm = (Bbe != nullptr) && (sBk == 1);
  bool anyTdm = aTdm || bTdm;
  unsigned ldsA[2] = { (unsigned)(unsigned long long)(void*)&As[0][0],
                       (unsigned)(unsigned long long)(void*)&As[1][0] };
  unsigned ldsB[2] = { (unsigned)(unsigned long long)(void*)&Bs[0][0],
                       (unsigned)(unsigned long long)(void*)&Bs[1][0] };

  auto vecLoadA = [&](int k0, uint4& d0, uint4& d1) {
    int r  = tid >> 1;
    int kk = (tid & 1) << 4;
    int gr = mTile + r;
    uint4 z = {0u, 0u, 0u, 0u};
    d0 = z; d1 = z;
    if (gr < cnt) {
      long sr = aRowIdx ? (long)aRowIdx[rowBase + gr] : (long)(rowBase + gr);
      const uint4* src = (const uint4*)(A + sr * sAm + k0 + kk);
      d0 = src[0];
      d1 = src[1];
    }
  };
  auto vecStoreA = [&](int buf, uint4 d0, uint4 d1) {
    int r  = tid >> 1;
    int kk = (tid & 1) << 4;
    *(uint4*)(&As[buf][r * BK + kk])     = d0;
    *(uint4*)(&As[buf][r * BK + kk + 8]) = d1;
  };
  auto vecLoadB = [&](int k0, float* nB) {
#pragma unroll
    for (int j = 0; j < 16; ++j) {
      int id = tid + 256 * j;
      int kq = id >> 7, n = id & 127;
      long so = (long)(k0 + kq) * sBk + (long)(nTile + n) * sBn;
      nB[j] = Bfe ? Bfe[so] : (float)Bbe[so];
    }
  };
  auto vecStoreB = [&](int buf, const float* nB) {
#pragma unroll
    for (int j = 0; j < 16; ++j) {
      int id = tid + 256 * j;
      int kq = id >> 7, n = id & 127;
      Bs[buf][n * BK + kq] = (bf16_t)nB[j];
    }
  };

  v8f acc[4][2];
#pragma unroll
  for (int i = 0; i < 4; ++i)
#pragma unroll
    for (int j = 0; j < 2; ++j)
#pragma unroll
      for (int v = 0; v < 8; ++v) acc[i][j][v] = 0.0f;

  // ---- prologue: stage K-step 0 into buffer 0 ----
  if (wave == 0) {
    if (aTdm) tdm_tile_128x32(A + (long)mTile * sAm, sAm, ldsA[0]);
    if (bTdm) tdm_tile_128x32(Bbe + (long)nTile * sBn, sBn, ldsB[0]);
  }
  if (!aTdm) { uint4 a0, a1; vecLoadA(0, a0, a1); vecStoreA(0, a0, a1); }
  if (!bTdm) { float nB[16]; vecLoadB(0, nB); vecStoreB(0, nB); }

  int nk = K / BK;
  for (int kt = 0; kt < nk; ++kt) {
    int cur = kt & 1;
    if (wave == 0 && anyTdm) __builtin_amdgcn_s_wait_tensorcnt(0);
    __syncthreads();

    // ---- fragments for current buffer (k = 16*(lane/16) + s, same order in A and B) ----
    union AF { v16bf v; unsigned u[8]; } af[4];
    union BF { v16bf v; unsigned u[8]; } bfr[2];
    const unsigned* Asu = (const unsigned*)&As[cur][0];
    const unsigned* Bsu = (const unsigned*)&Bs[cur][0];
#pragma unroll
    for (int i = 0; i < 4; ++i) {
      int row  = wm + 16 * i + lm;
      int base = row * (BK / 2);
#pragma unroll
      for (int p = 0; p < 8; ++p) {
        int kel = 2 * p + (aKsel << 1);
        af[i].u[p] = Asu[base + (kel >> 1)];
      }
    }
#pragma unroll
    for (int j = 0; j < 2; ++j) {
      int ncol = wn + 16 * j + lm;
      int base = ncol * (BK / 2) + (bKsel >> 1);
#pragma unroll
      for (int p = 0; p < 8; ++p) bfr[j].u[p] = Bsu[base + p];
    }

    // ---- issue next-tile staging before the WMMA block ----
    bool haveNext = (kt + 1 < nk);
    int nxt = cur ^ 1;
    int k0n = (kt + 1) * BK;
    uint4 nA0, nA1;
    float nB[16];
    if (haveNext) {
      if (wave == 0) {
        if (aTdm) tdm_tile_128x32(A + (long)mTile * sAm + k0n, sAm, ldsA[nxt]);
        if (bTdm) tdm_tile_128x32(Bbe + (long)nTile * sBn + k0n, sBn, ldsB[nxt]);
      }
      if (!aTdm) vecLoadA(k0n, nA0, nA1);
      if (!bTdm) {
        vecLoadB(k0n, nB);
        if (kt + 2 < nk) {   // prefetch K-step kt+2 (global_prefetch_b8)
          long pf = (long)(kt + 2) * BK * sBk + (long)(nTile + (tid & 127)) * sBn;
          if (Bfe) __builtin_prefetch((const void*)(Bfe + pf), 0, 1);
          else     __builtin_prefetch((const void*)(Bbe + pf), 0, 1);
        }
      }
    }

    // ---- 8 WMMAs (memory latency of staging hides behind these) ----
#pragma unroll
    for (int i = 0; i < 4; ++i)
#pragma unroll
      for (int j = 0; j < 2; ++j)
        acc[i][j] = __builtin_amdgcn_wmma_f32_16x16x32_bf16(
            false, af[i].v, false, bfr[j].v, (short)0, acc[i][j], false, false);

    // ---- drain staged registers into the next buffer ----
    if (haveNext) {
      if (!aTdm) vecStoreA(nxt, nA0, nA1);
      if (!bTdm) vecStoreB(nxt, nB);
    }
  }

  // ---- epilogue ----
#pragma unroll
  for (int i = 0; i < 4; ++i)
#pragma unroll
    for (int j = 0; j < 2; ++j)
#pragma unroll
      for (int v = 0; v < 8; ++v) {
        int ml = wm + 16 * i + ((lane >> 4) << 3) + v;
        int nl = wn + 16 * j + (lane & 15);
        int gr = mTile + ml;
        if (gr < cnt) {
          long r = rowBase + gr;
          float val = acc[i][j][v];
          if (epi == 0) {
            C[r * sCm + nTile + nl] = val;
          } else {
            int tok = epiTok[r];
            atomicAdd(outAcc + (long)tok * outStride + nTile + nl, val * rowGamma[r]);
          }
        }
      }
}

// ---------------------------------------------------------------------------
// RMSNorm (H=2048): writes bf16 (GEMM A operand) and optional f32 copy.
// ---------------------------------------------------------------------------
__global__ __launch_bounds__(256) void k_rmsnorm(const float* __restrict__ x,
                                                 const float* __restrict__ w,
                                                 bf16_t* __restrict__ ob,
                                                 float* __restrict__ of)
{
  int t = blockIdx.x, tid = threadIdx.x;
  const float* xr = x + (long)t * HH;
  float vb[8], ss = 0.f;
#pragma unroll
  for (int i = 0; i < 8; ++i) { float v = xr[tid + 256 * i]; vb[i] = v; ss += v * v; }
  ss = waveSum(ss);
  __shared__ float red[8];
  if ((tid & 31) == 0) red[tid >> 5] = ss;
  __syncthreads();
  float tot = 0.f;
#pragma unroll
  for (int i = 0; i < 8; ++i) tot += red[i];
  float sc = rsqrtf(tot / (float)HH + 1e-6f);
#pragma unroll
  for (int i = 0; i < 8; ++i) {
    int c = tid + 256 * i;
    float o = vb[i] * sc * w[c];
    ob[(long)t * HH + c] = (bf16_t)o;
    if (of) of[(long)t * HH + c] = o;
  }
}

// ---------------------------------------------------------------------------
// Fused per-head QK RMSNorm + RoPE (+ V convert). One wave per (token, head).
// ---------------------------------------------------------------------------
__global__ __launch_bounds__(256) void k_rope(const float* __restrict__ qkv,
                                              const int* __restrict__ pos,
                                              const float* __restrict__ qw,
                                              const float* __restrict__ kw,
                                              bf16_t* __restrict__ qout,
                                              bf16_t* __restrict__ kout,
                                              bf16_t* __restrict__ vout)
{
  int w = (blockIdx.x * 256 + threadIdx.x) >> 5;
  int lane = threadIdx.x & 31;
  int t = w / 40, task = w % 40;
  if (t >= TT) return;
  const float* row = qkv + (long)t * QKVN;

  if (task < 36) {
    bool isQ = task < 32;
    int head = isQ ? task : (task - 32);
    const float* x = isQ ? (row + head * HDIM) : (row + NHD * HDIM + head * HDIM);
    const float* nw = isQ ? qw : kw;
    float a0 = x[lane], a1 = x[lane + 32], b0 = x[lane + 64], b1 = x[lane + 96];
    float ss = waveSum(a0 * a0 + a1 * a1 + b0 * b0 + b1 * b1);
    float sc = rsqrtf(ss / (float)HDIM + 1e-6f);
    a0 *= sc * nw[lane];      a1 *= sc * nw[lane + 32];
    b0 *= sc * nw[lane + 64]; b1 *= sc * nw[lane + 96];
    float p = (float)pos[t];
    float f0 = powf(10000.f, -((float)lane) / 64.f);
    float f1 = powf(10000.f, -((float)(lane + 32)) / 64.f);
    float c0 = cosf(p * f0), s0 = sinf(p * f0);
    float c1 = cosf(p * f1), s1 = sinf(p * f1);
    bf16_t* o = isQ ? (qout + ((long)t * NHD + head) * HDIM)
                    : (kout + ((long)t * NKVH + head) * HDIM);
    o[lane]      = (bf16_t)(a0 * c0 - b0 * s0);
    o[lane + 32] = (bf16_t)(a1 * c1 - b1 * s1);
    o[lane + 64] = (bf16_t)(b0 * c0 + a0 * s0);
    o[lane + 96] = (bf16_t)(b1 * c1 + a1 * s1);
  } else {
    int vh = task - 36;
    const float* x = row + (NHD + NKVH) * HDIM + vh * HDIM;
    bf16_t* o = vout + ((long)t * NKVH + vh) * HDIM;
    o[lane]      = (bf16_t)x[lane];
    o[lane + 32] = (bf16_t)x[lane + 32];
    o[lane + 64] = (bf16_t)x[lane + 64];
    o[lane + 96] = (bf16_t)x[lane + 96];
  }
}

// ---------------------------------------------------------------------------
// Causal softmax over one score row; writes P as bf16 (PV GEMM A operand).
// ---------------------------------------------------------------------------
__global__ __launch_bounds__(256) void k_softmax(const float* __restrict__ S,
                                                 bf16_t* __restrict__ P)
{
  int q = blockIdx.x, tid = threadIdx.x;
  const float scale = 0.08838834764831845f;   // 1/sqrt(128)
  const float* row = S + (long)q * TT;
  int L = q + 1;
  float vals[8], m = -3.0e38f;
#pragma unroll
  for (int i = 0; i < 8; ++i) {
    int k = tid + 256 * i;
    float v = (k < L) ? row[k] * scale : -3.0e38f;
    vals[i] = v;
    m = fmaxf(m, v);
  }
  m = waveMax(m);
  __shared__ float red[8];
  if ((tid & 31) == 0) red[tid >> 5] = m;
  __syncthreads();
  float gm = -3.0e38f;
#pragma unroll
  for (int i = 0; i < 8; ++i) gm = fmaxf(gm, red[i]);
  __syncthreads();
  float s = 0.f;
#pragma unroll
  for (int i = 0; i < 8; ++i) {
    int k = tid + 256 * i;
    float ev = (k < L) ? __expf(vals[i] - gm) : 0.f;
    vals[i] = ev;
    s += ev;
  }
  s = waveSum(s);
  if ((tid & 31) == 0) red[tid >> 5] = s;
  __syncthreads();
  float tot = 0.f;
#pragma unroll
  for (int i = 0; i < 8; ++i) tot += red[i];
  float inv = 1.f / tot;
#pragma unroll
  for (int i = 0; i < 8; ++i) {
    int k = tid + 256 * i;
    P[(long)q * TT + k] = (bf16_t)(vals[i] * inv);
  }
}

// ---------------------------------------------------------------------------
// Router: gate logits (one wave per (token, expert)).
// ---------------------------------------------------------------------------
__global__ __launch_bounds__(256) void k_gate(const float* __restrict__ xn,
                                              const float* __restrict__ gw,
                                              float* __restrict__ logits)
{
  int w = (blockIdx.x * 256 + threadIdx.x) >> 5;
  int lane = threadIdx.x & 31;
  int t = w >> 6, e = w & 63;
  if (t >= TT) return;
  const float* xr = xn + (long)t * HH;
  float s = 0.f;
  for (int i = lane; i < HH; i += 32) s += xr[i] * gw[(long)i * NEXP + e];
  s = waveSum(s);
  if (lane == 0) logits[(long)t * NEXP + e] = s;
}

// ---------------------------------------------------------------------------
// Top-8 over 64 softmax probs; one wave per token; also counts rows/expert.
// ---------------------------------------------------------------------------
__global__ __launch_bounds__(256) void k_topk(const float* __restrict__ logits,
                                              int* __restrict__ idxOut,
                                              float* __restrict__ gamOut,
                                              int* __restrict__ cnt)
{
  int t = (blockIdx.x * 256 + threadIdx.x) >> 5;
  int lane = threadIdx.x & 31;
  if (t >= TT) return;
  float v0 = logits[(long)t * NEXP + lane];
  float v1 = logits[(long)t * NEXP + 32 + lane];
  float m = waveMax(fmaxf(v0, v1));
  float p0 = __expf(v0 - m), p1 = __expf(v1 - m);
  float s = waveSum(p0 + p1);
  p0 /= s; p1 /= s;
  int selI[TOPKK]; float selV[TOPKK];
#pragma unroll
  for (int r = 0; r < TOPKK; ++r) {
    float bv; int bi;
    if (p0 >= p1) { bv = p0; bi = lane; } else { bv = p1; bi = lane + 32; }
#pragma unroll
    for (int o = 16; o > 0; o >>= 1) {
      float ov = __shfl_xor(bv, o, 32);
      int   oi = __shfl_xor(bi, o, 32);
      if (ov > bv || (ov == bv && oi < bi)) { bv = ov; bi = oi; }
    }
    selI[r] = bi; selV[r] = bv;
    if (bi == lane)      p0 = -1.f;
    if (bi == lane + 32) p1 = -1.f;
  }
  float st = 0.f;
#pragma unroll
  for (int r = 0; r < TOPKK; ++r) st += selV[r];
  if (lane == 0) {
#pragma unroll
    for (int r = 0; r < TOPKK; ++r) {
      idxOut[(long)t * TOPKK + r] = selI[r];
      gamOut[(long)t * TOPKK + r] = selV[r] / st;
      atomicAdd(&cnt[selI[r]], 1);
    }
  }
}

__global__ void k_scan(const int* __restrict__ cnt, int* __restrict__ goff) {
  if (threadIdx.x == 0 && blockIdx.x == 0) {
    int s = 0;
    for (int e = 0; e < NEXP; ++e) { goff[e] = s; s += cnt[e]; }
  }
}

__global__ __launch_bounds__(256) void k_scatter(const int* __restrict__ idx,
                                                 const float* __restrict__ gam,
                                                 const int* __restrict__ goff,
                                                 int* __restrict__ cur,
                                                 int* __restrict__ rowTok,
                                                 float* __restrict__ rowGam)
{
  int i = blockIdx.x * 256 + threadIdx.x;
  if (i >= NROWS) return;
  int e = idx[i];
  int p = goff[e] + atomicAdd(&cur[e], 1);
  rowTok[p] = i >> 3;
  rowGam[p] = gam[i];
}

// ---------------------------------------------------------------------------
// small elementwise helpers
// ---------------------------------------------------------------------------
__global__ __launch_bounds__(256) void k_cvt(const float* __restrict__ x,
                                             bf16_t* __restrict__ y, long n) {
  long i = (long)blockIdx.x * 256 + threadIdx.x;
  if (i < n) y[i] = (bf16_t)x[i];
}
__global__ __launch_bounds__(256) void k_add(const float* __restrict__ a,
                                             const float* __restrict__ b,
                                             float* __restrict__ y, long n) {
  long i = (long)blockIdx.x * 256 + threadIdx.x;
  if (i < n) y[i] = a[i] + b[i];
}
__global__ __launch_bounds__(256) void k_copy(const float* __restrict__ a,
                                              float* __restrict__ y, long n) {
  long i = (long)blockIdx.x * 256 + threadIdx.x;
  if (i < n) y[i] = a[i];
}
__global__ __launch_bounds__(256) void k_zero(int* __restrict__ p, int n) {
  int i = blockIdx.x * blockDim.x + threadIdx.x;
  if (i < n) p[i] = 0;
}
__global__ __launch_bounds__(256) void k_silu(const float* __restrict__ h,
                                              bf16_t* __restrict__ a) {
  long i = (long)blockIdx.x * 256 + threadIdx.x;
  if (i >= (long)NROWS * II) return;
  long r = i / II, c = i % II;
  float g = h[r * (2 * II) + c];
  float u = h[r * (2 * II) + II + c];
  float act = (g / (1.f + __expf(-g))) * u;
  a[i] = (bf16_t)act;
}

// ---------------------------------------------------------------------------
extern "C" void kernel_launch(void* const* d_in, const int* in_sizes, int n_in,
                              void* d_out, int out_size, void* d_ws, size_t ws_size,
                              hipStream_t stream)
{
  (void)in_sizes; (void)n_in; (void)out_size; (void)ws_size;
  const float* hs   = (const float*)d_in[0];
  const int*   pos  = (const int*)d_in[1];
  const float* wqkv = (const float*)d_in[2];
  const float* qnw  = (const float*)d_in[3];
  const float* knw  = (const float*)d_in[4];
  const float* wo   = (const float*)d_in[5];
  const float* ln1  = (const float*)d_in[6];
  const float* ln2  = (const float*)d_in[7];
  const float* gw   = (const float*)d_in[8];
  const float* wgu  = (const float*)d_in[9];
  const float* wdn  = (const float*)d_in[10];
  float* out = (float*)d_out;

  char* base = (char*)d_ws;
  size_t off = 0;
  auto alloc = [&](size_t bytes) -> char* {
    char* p = base + off;
    off = (off + bytes + 255) & ~(size_t)255;
    return p;
  };
  bf16_t* xnorm  = (bf16_t*)alloc((size_t)TT * HH * 2);
  float*  qkv    = (float*) alloc((size_t)TT * QKVN * 4);
  bf16_t* qb     = (bf16_t*)alloc((size_t)TT * NHD  * HDIM * 2);
  bf16_t* kb     = (bf16_t*)alloc((size_t)TT * NKVH * HDIM * 2);
  bf16_t* vb     = (bf16_t*)alloc((size_t)TT * NKVH * HDIM * 2);
  float*  sc     = (float*) alloc((size_t)TT * TT * 4);
  bf16_t* Pb     = (bf16_t*)alloc((size_t)TT * TT * 2);
  float*  aout   = (float*) alloc((size_t)TT * NHD * HDIM * 4);
  bf16_t* aoutb  = (bf16_t*)alloc((size_t)TT * NHD * HDIM * 2);
  float*  aproj  = (float*) alloc((size_t)TT * HH * 4);
  float*  h2     = (float*) alloc((size_t)TT * HH * 4);
  bf16_t* h2nb   = (bf16_t*)alloc((size_t)TT * HH * 2);
  float*  h2nf   = (float*) alloc((size_t)TT * HH * 4);
  float*  lg     = (float*) alloc((size_t)TT * NEXP * 4);
  int*    tki    = (int*)   alloc((size_t)TT * TOPKK * 4);
  float*  tkg    = (float*) alloc((size_t)TT * TOPKK * 4);
  int*    cnt    = (int*)   alloc(128 * 4);        // cnt[64] + cursor[64]
  int*    cur    = cnt + NEXP;
  int*    goff   = (int*)   alloc(NEXP * 4);
  int*    rowTok = (int*)   alloc((size_t)NROWS * 4);
  float*  rowGam = (float*) alloc((size_t)NROWS * 4);
  float*  moeh   = (float*) alloc((size_t)NROWS * 2 * II * 4);
  bf16_t* ab     = (bf16_t*)alloc((size_t)NROWS * II * 2);

  // ---- LN1 -> bf16, QKV projection ----
  k_rmsnorm<<<TT, 256, 0, stream>>>(hs, ln1, xnorm, (float*)nullptr);
  k_gemm<<<dim3(QKVN / BN, TT / BM, 1), 256, 0, stream>>>(
      xnorm, HH, wqkv, nullptr, QKVN, 1, 0, qkv, QKVN,
      TT, QKVN, HH, nullptr, nullptr, nullptr, nullptr, nullptr, nullptr, 0, 0);
  k_rope<<<(TT * 40) / 8, 256, 0, stream>>>(qkv, pos, qnw, knw, qb, kb, vb);

  // ---- attention, head by head (score buffer fits in L2) ----
  for (int h = 0; h < NHD; ++h) {
    int kvh = h / (NHD / NKVH);
    k_gemm<<<dim3(TT / BN, TT / BM, 1), 256, 0, stream>>>(
        qb + h * HDIM, (long)NHD * HDIM, nullptr, kb + kvh * HDIM,
        1, (long)NKVH * HDIM, 0, sc, TT,
        TT, TT, HDIM, nullptr, nullptr, nullptr, nullptr, nullptr, nullptr, 0, 0);
    k_softmax<<<TT, 256, 0, stream>>>(sc, Pb);
    k_gemm<<<dim3(HDIM / BN, TT / BM, 1), 256, 0, stream>>>(
        Pb, TT, nullptr, vb + kvh * HDIM,
        (long)NKVH * HDIM, 1, 0, aout + h * HDIM, (long)NHD * HDIM,
        TT, HDIM, TT, nullptr, nullptr, nullptr, nullptr, nullptr, nullptr, 0, 0);
  }
  k_cvt<<<(TT * NHD * HDIM) / 256, 256, 0, stream>>>(aout, aoutb, (long)TT * NHD * HDIM);
  k_gemm<<<dim3(HH / BN, TT / BM, 1), 256, 0, stream>>>(
      aoutb, (long)NHD * HDIM, wo, nullptr, HH, 1, 0, aproj, HH,
      TT, HH, NHD * HDIM, nullptr, nullptr, nullptr, nullptr, nullptr, nullptr, 0, 0);
  k_add<<<(TT * HH) / 256, 256, 0, stream>>>(hs, aproj, h2, (long)TT * HH);

  // ---- LN2 + router + ragged MoE ----
  k_rmsnorm<<<TT, 256, 0, stream>>>(h2, ln2, h2nb, h2nf);
  k_gate<<<(TT * NEXP) / 8, 256, 0, stream>>>(h2nf, gw, lg);
  k_zero<<<1, 128, 0, stream>>>(cnt, 128);
  k_topk<<<TT / 8, 256, 0, stream>>>(lg, tki, tkg, cnt);
  k_scan<<<1, 1, 0, stream>>>(cnt, goff);
  k_scatter<<<NROWS / 256, 256, 0, stream>>>(tki, tkg, goff, cur, rowTok, rowGam);

  k_copy<<<(TT * HH) / 256, 256, 0, stream>>>(h2, out, (long)TT * HH);

  // gate_up: ragged per-expert GEMM, A rows gathered via rowTok
  k_gemm<<<dim3((2 * II) / BN, 128, NEXP), 256, 0, stream>>>(
      h2nb, HH, wgu, nullptr, 2 * II, 1, (long)HH * 2 * II,
      moeh, 2 * II, NROWS, 2 * II, HH,
      rowTok, goff, cnt, nullptr, nullptr, nullptr, 0, 0);
  k_silu<<<((long)NROWS * II) / 256, 256, 0, stream>>>(moeh, ab);
  // down: ragged GEMM, epilogue does gamma-scaled atomicAdd scatter into out
  k_gemm<<<dim3(HH / BN, 128, NEXP), 256, 0, stream>>>(
      ab, II, wdn, nullptr, HH, 1, (long)II * HH,
      moeh /*unused*/, 0, NROWS, HH, II,
      nullptr, goff, cnt, rowGam, rowTok, out, HH, 1);
}
